// DependencyParserCombinedAttention_62010737819977
// MI455X (gfx1250) — matrix operand order repaired and checked by reading, not compiled, not run on hardware
//
#include <hip/hip_runtime.h>
#include <hip/hip_bf16.h>
#include <math.h>

// ---- shapes ----
#define SEQ_N 768
#define D0P   384     // E_W+E_P = 364 padded to mult of 32
#define D1    1024    // 2H
#define G4H   2048    // 4H
#define HID   512
#define MD    256
#define PADM  320     // M+1 = 257 padded to mult of 64 (N-tile) and 32 (K-step)

typedef _Float16 half8  __attribute__((ext_vector_type(8)));
typedef _Float16 half16 __attribute__((ext_vector_type(16)));
typedef float    float8 __attribute__((ext_vector_type(8)));

// ---------------------------------------------------------------------------
// WMMA GEMM:  C[m,n] = sum_k A[m,k] * W[n,k]  (+ bias[n])
// A: row-major f16 [M x lda], W: row-major f16 [N x ldw] (i.e. B = W^T).
// One wave (32 threads) computes a 16(M) x 64(N) strip: a single A fragment
// is amortized over 4 B fragments / 4 accumulators per K-step (K mult of 32).
// Stores f16 to Ch if non-null, else f32 to Cf.
// Fragment layouts used here (A/B: contiguous k order, same for both; D as in ISA):
//  A frag (16x32 f16): lane l: row = l&15; halfs[0..15]=K 16*(l>>4)..+15
//  B frag (32x16 f16): lane l: col = l&15; halfs[0..15]=K 16*(l>>4)..+15
//  D frag (16x16 f32): lane l: col = l&15; vgpr r: row = r + 8*(l>>4)
// ---------------------------------------------------------------------------
__global__ void gemm_f16_wmma(const _Float16* __restrict__ A, int lda,
                              const _Float16* __restrict__ W, int ldw,
                              float* __restrict__ Cf, _Float16* __restrict__ Ch,
                              int ldc, const float* __restrict__ bias, int K)
{
    const int lane = threadIdx.x & 31;
    const int l15  = lane & 15;
    const int lhi  = lane >> 4;
    const int nTile = blockIdx.x * 64;
    const int mTile = blockIdx.y * 16;

    const _Float16* Arow = A + (size_t)(mTile + l15) * lda;
    const _Float16* Wrow0 = W + (size_t)(nTile +  0 + l15) * ldw;
    const _Float16* Wrow1 = W + (size_t)(nTile + 16 + l15) * ldw;
    const _Float16* Wrow2 = W + (size_t)(nTile + 32 + l15) * ldw;
    const _Float16* Wrow3 = W + (size_t)(nTile + 48 + l15) * ldw;

    float8 acc0 = {0.f,0.f,0.f,0.f,0.f,0.f,0.f,0.f};
    float8 acc1 = acc0, acc2 = acc0, acc3 = acc0;

    for (int k = 0; k < K; k += 32) {
        // A fragment: one contiguous 32B run
        half8 alo = *(const half8*)(Arow + k + lhi * 16);
        half8 ahi = *(const half8*)(Arow + k + lhi * 16 + 8);
        // Four B fragments: one contiguous 32B run each
        half8 b0l = *(const half8*)(Wrow0 + k + lhi * 16);
        half8 b0h = *(const half8*)(Wrow0 + k + lhi * 16 + 8);
        half8 b1l = *(const half8*)(Wrow1 + k + lhi * 16);
        half8 b1h = *(const half8*)(Wrow1 + k + lhi * 16 + 8);
        half8 b2l = *(const half8*)(Wrow2 + k + lhi * 16);
        half8 b2h = *(const half8*)(Wrow2 + k + lhi * 16 + 8);
        half8 b3l = *(const half8*)(Wrow3 + k + lhi * 16);
        half8 b3h = *(const half8*)(Wrow3 + k + lhi * 16 + 8);
        if (k + 32 < K) {                       // global_prefetch_b8 next K-slab
            __builtin_prefetch(Arow  + k + 32, 0, 1);
            __builtin_prefetch(Wrow0 + k + 32, 0, 1);
            __builtin_prefetch(Wrow1 + k + 32, 0, 1);
            __builtin_prefetch(Wrow2 + k + 32, 0, 1);
            __builtin_prefetch(Wrow3 + k + 32, 0, 1);
        }
        half16 a, b0, b1, b2, b3;
#pragma unroll
        for (int i = 0; i < 8; ++i) {
            a[i]  = alo[i]; a[i + 8]  = ahi[i];
            b0[i] = b0l[i]; b0[i + 8] = b0h[i];
            b1[i] = b1l[i]; b1[i + 8] = b1h[i];
            b2[i] = b2l[i]; b2[i + 8] = b2h[i];
            b3[i] = b3l[i]; b3[i + 8] = b3h[i];
        }
        acc0 = __builtin_amdgcn_wmma_f32_16x16x32_f16(false, a, false, b0, (short)0, acc0, false, false);
        acc1 = __builtin_amdgcn_wmma_f32_16x16x32_f16(false, a, false, b1, (short)0, acc1, false, false);
        acc2 = __builtin_amdgcn_wmma_f32_16x16x32_f16(false, a, false, b2, (short)0, acc2, false, false);
        acc3 = __builtin_amdgcn_wmma_f32_16x16x32_f16(false, a, false, b3, (short)0, acc3, false, false);
    }

    const int m0 = mTile + lhi * 8;
    float8 accs[4] = {acc0, acc1, acc2, acc3};
#pragma unroll
    for (int c = 0; c < 4; ++c) {
        const int n = nTile + 16 * c + l15;
        const float bv = bias ? bias[n] : 0.f;
#pragma unroll
        for (int r = 0; r < 8; ++r) {
            float v = accs[c][r] + bv;
            if (Ch) Ch[(size_t)(m0 + r) * ldc + n] = (_Float16)v;
            else    Cf[(size_t)(m0 + r) * ldc + n] = v;
        }
    }
}

// ---------------------------------------------------------------------------
// Embedding gather -> f16, padded to D0P columns with zeros.
// ---------------------------------------------------------------------------
__global__ void embed_kernel(const int* __restrict__ widx, const int* __restrict__ pidx,
                             const float* __restrict__ wemb, const float* __restrict__ pemb,
                             _Float16* __restrict__ x0h)
{
    const int row = blockIdx.x;
    const int w = widx[row], p = pidx[row];
    for (int k = threadIdx.x; k < D0P; k += blockDim.x) {
        float v = 0.f;
        if (k < 300)      v = wemb[(size_t)w * 300 + k];
        else if (k < 364) v = pemb[(size_t)p * 64 + (k - 300)];
        x0h[(size_t)row * D0P + k] = (_Float16)v;
    }
}

// f32 [N x K] -> f16 [N x Kp] zero-padded rows
__global__ void cvt_pad_f16(const float* __restrict__ src, _Float16* __restrict__ dst,
                            int Nrows, int K, int Kp)
{
    int idx = blockIdx.x * blockDim.x + threadIdx.x;
    int total = Nrows * Kp;
    if (idx >= total) return;
    int rr = idx / Kp, kk = idx - rr * Kp;
    dst[idx] = (kk < K) ? (_Float16)src[(size_t)rr * K + kk] : (_Float16)0.f;
}

// ATh[n,k] = A[k,n] (257x257), zero-padded to PADM x PADM
__global__ void build_at(const float* __restrict__ A, _Float16* __restrict__ dst)
{
    int idx = blockIdx.x * blockDim.x + threadIdx.x;
    if (idx >= PADM * PADM) return;
    int nn = idx / PADM, kk = idx - nn * PADM;
    dst[idx] = (nn < 257 && kk < 257) ? (_Float16)A[(size_t)kk * 257 + nn]
                                      : (_Float16)0.f;
}

// dst[row, 0..255] = tanh(src[row, 0..255]); dst[row,256]=1; dst[row,257..]=0
__global__ void tanh_append(const float* __restrict__ src, _Float16* __restrict__ dst)
{
    int idx = blockIdx.x * blockDim.x + threadIdx.x;
    if (idx >= SEQ_N * PADM) return;
    int rr = idx / PADM, kk = idx - rr * PADM;
    float v;
    if (kk < MD)       v = tanhf(src[(size_t)rr * MD + kk]);
    else if (kk == MD) v = 1.f;
    else               v = 0.f;
    dst[idx] = (_Float16)v;
}

// ---------------------------------------------------------------------------
// One BiLSTM layer direction-pair. Gates G = x@Wih^T + b precomputed.
// blockIdx.x = 0 -> forward, 1 -> backward. 512 threads, h kept in LDS.
// Writes f16 h into Xout[row, dir*512 + tid] (row-major ld = 1024).
// ---------------------------------------------------------------------------
__global__ void lstm_dir_kernel(const float* __restrict__ Gf, const float* __restrict__ Gb,
                                const float* __restrict__ Whhf, const float* __restrict__ Whhb,
                                _Float16* __restrict__ Xout)
{
    const int dir = blockIdx.x;
    const float* __restrict__ G   = dir ? Gb   : Gf;
    const float* __restrict__ Whh = dir ? Whhb : Whhf;
    const int tid = threadIdx.x;          // 0..511

    __shared__ float hsh[HID];
    hsh[tid] = 0.f;
    float c = 0.f;
    __syncthreads();

    const float* w_i = Whh + (size_t)tid * HID;
    const float* w_f = Whh + (size_t)(HID  + tid) * HID;
    const float* w_g = Whh + (size_t)(2*HID + tid) * HID;
    const float* w_o = Whh + (size_t)(3*HID + tid) * HID;

    for (int t = 0; t < SEQ_N; ++t) {
        const int row = dir ? (SEQ_N - 1 - t) : t;
        const float* g = G + (size_t)row * G4H;
        float zi = g[tid], zf = g[HID + tid], zg = g[2*HID + tid], zo = g[3*HID + tid];
#pragma unroll 8
        for (int k = 0; k < HID; ++k) {
            float hv = hsh[k];
            zi = fmaf(w_i[k], hv, zi);
            zf = fmaf(w_f[k], hv, zf);
            zg = fmaf(w_g[k], hv, zg);
            zo = fmaf(w_o[k], hv, zo);
        }
        float ig = 1.f / (1.f + __expf(-zi));
        float fg = 1.f / (1.f + __expf(-zf));
        float og = 1.f / (1.f + __expf(-zo));
        c = fg * c + ig * tanhf(zg);
        float hn = og * tanhf(c);
        __syncthreads();                 // all dots on old h done
        hsh[tid] = hn;
        Xout[(size_t)row * D1 + dir * HID + tid] = (_Float16)hn;
        __syncthreads();
    }
}

// ---------------------------------------------------------------------------
// Pairwise MLP score: out[i,j] += bf + sum_k tanh(h[i,k]+m[j,k]) * Wf[k]
// 32x32 output tile per block, k chunked through LDS (KC=64).
// ---------------------------------------------------------------------------
#define KC 64
__global__ void pair_mlp_kernel(const float* __restrict__ hbuf, const float* __restrict__ mbuf,
                                const float* __restrict__ Wf, const float* __restrict__ bf,
                                float* __restrict__ out)
{
    __shared__ float hs[32][KC];
    __shared__ float ms[32][KC];
    __shared__ float wf[KC];

    const int i0 = blockIdx.y * 32;
    const int j0 = blockIdx.x * 32;
    const int tid = threadIdx.x;          // 256 threads
    const int tj  = tid & 31;
    const int tiq = tid >> 5;             // 0..7

    float acc[4] = {0.f, 0.f, 0.f, 0.f};

    for (int kc = 0; kc < MD; kc += KC) {
        __syncthreads();
#pragma unroll
        for (int e = tid; e < 32 * KC; e += 256) {
            int rr = e / KC, kk = e - rr * KC;
            hs[rr][kk] = hbuf[(size_t)(i0 + rr) * MD + kc + kk];
            ms[rr][kk] = mbuf[(size_t)(j0 + rr) * MD + kc + kk];
        }
        if (tid < KC) wf[tid] = Wf[kc + tid];
        __syncthreads();
#pragma unroll 4
        for (int kk = 0; kk < KC; ++kk) {
            float mv = ms[tj][kk];
            float wv = wf[kk];
#pragma unroll
            for (int q = 0; q < 4; ++q)
                acc[q] += tanhf(hs[tiq + 8 * q][kk] + mv) * wv;
        }
    }
    const float b = bf[0];
#pragma unroll
    for (int q = 0; q < 4; ++q) {
        int i = i0 + tiq + 8 * q;
        out[(size_t)i * SEQ_N + j0 + tj] += acc[q] + b;
    }
}

// ---------------------------------------------------------------------------
extern "C" void kernel_launch(void* const* d_in, const int* in_sizes, int n_in,
                              void* d_out, int out_size, void* d_ws, size_t ws_size,
                              hipStream_t stream)
{
    const int*   widx  = (const int*)  d_in[0];
    const int*   pidx  = (const int*)  d_in[1];
    const float* wemb  = (const float*)d_in[2];
    const float* pemb  = (const float*)d_in[3];
    const float* Wih0f = (const float*)d_in[4];
    const float* Whh0f = (const float*)d_in[5];
    const float* b0f   = (const float*)d_in[6];
    const float* Wih0b = (const float*)d_in[7];
    const float* Whh0b = (const float*)d_in[8];
    const float* b0b   = (const float*)d_in[9];
    const float* Wih1f = (const float*)d_in[10];
    const float* Whh1f = (const float*)d_in[11];
    const float* b1f   = (const float*)d_in[12];
    const float* Wih1b = (const float*)d_in[13];
    const float* Whh1b = (const float*)d_in[14];
    const float* b1b   = (const float*)d_in[15];
    const float* Wh    = (const float*)d_in[16];
    const float* bh    = (const float*)d_in[17];
    const float* Wm    = (const float*)d_in[18];
    const float* bm    = (const float*)d_in[19];
    const float* Amat  = (const float*)d_in[20];
    const float* Wf    = (const float*)d_in[21];
    const float* bfp   = (const float*)d_in[22];
    float* out = (float*)d_out;

    // ---- carve workspace (~33 MB) ----
    char* wsp = (char*)d_ws;
    auto carve = [&](size_t bytes) -> void* {
        void* p = (void*)wsp;
        wsp += (bytes + 255) & ~(size_t)255;
        return p;
    };
    _Float16* x0h  = (_Float16*)carve((size_t)SEQ_N * D0P * 2);
    _Float16* w0fh = (_Float16*)carve((size_t)G4H * D0P * 2);
    _Float16* w0bh = (_Float16*)carve((size_t)G4H * D0P * 2);
    _Float16* x1h  = (_Float16*)carve((size_t)SEQ_N * D1 * 2);
    _Float16* w1fh = (_Float16*)carve((size_t)G4H * D1 * 2);
    _Float16* w1bh = (_Float16*)carve((size_t)G4H * D1 * 2);
    _Float16* x2h  = (_Float16*)carve((size_t)SEQ_N * D1 * 2);
    _Float16* whH  = (_Float16*)carve((size_t)MD * D1 * 2);
    _Float16* wmH  = (_Float16*)carve((size_t)MD * D1 * 2);
    _Float16* hbh  = (_Float16*)carve((size_t)SEQ_N * PADM * 2);
    _Float16* mbh  = (_Float16*)carve((size_t)SEQ_N * PADM * 2);
    _Float16* ath  = (_Float16*)carve((size_t)PADM * PADM * 2);
    _Float16* th   = (_Float16*)carve((size_t)SEQ_N * PADM * 2);
    float* gF = (float*)carve((size_t)SEQ_N * G4H * 4);   // reused layer0/layer1
    float* gB = (float*)carve((size_t)SEQ_N * G4H * 4);
    float* hbuf = (float*)carve((size_t)SEQ_N * MD * 4);
    float* mbuf = (float*)carve((size_t)SEQ_N * MD * 4);
    (void)ws_size; (void)n_in; (void)in_sizes; (void)out_size;

    auto gemm = [&](const _Float16* A, int lda, const _Float16* W, int ldw,
                    float* Cf, _Float16* Ch, int ldc, const float* bias,
                    int Mr, int Nc, int K) {
        dim3 g(Nc / 64, Mr / 16);             // 16x64 strip per wave
        gemm_f16_wmma<<<g, 32, 0, stream>>>(A, lda, W, ldw, Cf, Ch, ldc, bias, K);
    };
    auto cvt = [&](const float* s, _Float16* d, int Nr, int K, int Kp) {
        int tot = Nr * Kp;
        cvt_pad_f16<<<(tot + 255) / 256, 256, 0, stream>>>(s, d, Nr, K, Kp);
    };

    // 1) weight conversions to f16 (padded K)
    cvt(Wih0f, w0fh, G4H, 364, D0P);
    cvt(Wih0b, w0bh, G4H, 364, D0P);
    cvt(Wih1f, w1fh, G4H, D1, D1);
    cvt(Wih1b, w1bh, G4H, D1, D1);
    cvt(Wh,    whH,  MD,  D1, D1);
    cvt(Wm,    wmH,  MD,  D1, D1);
    build_at<<<(PADM * PADM + 255) / 256, 256, 0, stream>>>(Amat, ath);

    // 2) embedding gather
    embed_kernel<<<SEQ_N, 128, 0, stream>>>(widx, pidx, wemb, pemb, x0h);

    // 3) layer 0: gate preactivations (WMMA), then recurrence
    gemm(x0h, D0P, w0fh, D0P, gF, nullptr, G4H, b0f, SEQ_N, G4H, D0P);
    gemm(x0h, D0P, w0bh, D0P, gB, nullptr, G4H, b0b, SEQ_N, G4H, D0P);
    lstm_dir_kernel<<<2, HID, 0, stream>>>(gF, gB, Whh0f, Whh0b, x1h);

    // 4) layer 1
    gemm(x1h, D1, w1fh, D1, gF, nullptr, G4H, b1f, SEQ_N, G4H, D1);
    gemm(x1h, D1, w1bh, D1, gB, nullptr, G4H, b1b, SEQ_N, G4H, D1);
    lstm_dir_kernel<<<2, HID, 0, stream>>>(gF, gB, Whh1f, Whh1b, x2h);

    // 5) projections h, m
    gemm(x2h, D1, whH, D1, hbuf, nullptr, MD, bh, SEQ_N, MD, D1);
    gemm(x2h, D1, wmH, D1, mbuf, nullptr, MD, bm, SEQ_N, MD, D1);

    // 6) biaffine attention: hb_=[tanh(h),1], mb_=[tanh(m),1]
    tanh_append<<<(SEQ_N * PADM + 255) / 256, 256, 0, stream>>>(hbuf, hbh);
    tanh_append<<<(SEQ_N * PADM + 255) / 256, 256, 0, stream>>>(mbuf, mbh);
    // T = hb_ @ A   (W = A^T so C[m,n] = sum_k hb[m,k]*A[k,n])
    gemm(hbh, PADM, ath, PADM, nullptr, th, PADM, nullptr, SEQ_N, PADM, PADM);
    // s_att = T @ mb_^T  -> d_out
    gemm(th, PADM, mbh, PADM, out, nullptr, SEQ_N, nullptr, SEQ_N, SEQ_N, PADM);

    // 7) pairwise tanh MLP, accumulated into d_out
    dim3 pg(SEQ_N / 32, SEQ_N / 32);
    pair_mlp_kernel<<<pg, 256, 0, stream>>>(hbuf, mbuf, Wf, bfp, out);
}
